// GNNModel_13365938226086
// MI455X (gfx1250) — matrix operand order repaired and checked by reading, not compiled, not run on hardware
//
#include <hip/hip_runtime.h>
#include <hip/hip_bf16.h>
#include <math.h>

typedef __attribute__((ext_vector_type(16))) _Float16 v16h;
typedef __attribute__((ext_vector_type(8)))  _Float16 v8h;
typedef __attribute__((ext_vector_type(8)))  float    v8f;

#define NEG_SLOPE 0.2f
#define LN_EPS 1e-5f

__device__ __forceinline__ unsigned f2ord(float f) {
  unsigned b = __float_as_uint(f);
  return (b & 0x80000000u) ? ~b : (b | 0x80000000u);
}
__device__ __forceinline__ float ord2f(unsigned u) {
  unsigned b = (u & 0x80000000u) ? (u & 0x7FFFFFFFu) : ~u;
  return __uint_as_float(b);
}

// ---------------- fills / converts ----------------
__global__ void fill_f32(float* p, long long n, float v) {
  long long t = (long long)blockIdx.x * blockDim.x + threadIdx.x;
  if (t < n) p[t] = v;
}
__global__ void fill_u32(unsigned* p, long long n, unsigned v) {
  long long t = (long long)blockIdx.x * blockDim.x + threadIdx.x;
  if (t < n) p[t] = v;
}
// elementwise f32 -> f16, 4 per thread
__global__ void cvt_f16(const float* __restrict__ in, _Float16* __restrict__ out,
                        long long n) {
  long long i = ((long long)blockIdx.x * blockDim.x + threadIdx.x) * 4;
  if (i + 3 < n) {
    const float4 v = *(const float4*)(in + i);
    out[i]     = (_Float16)v.x;
    out[i + 1] = (_Float16)v.y;
    out[i + 2] = (_Float16)v.z;
    out[i + 3] = (_Float16)v.w;
  } else {
    for (; i < n; ++i) out[i] = (_Float16)in[i];
  }
}
// W[k][c] (f32, KxN) -> Wt[c][k] (f16, NxK)
__global__ void cvt_transpose_f16(const float* __restrict__ W,
                                  _Float16* __restrict__ Wt, int K, int Ncols) {
  int t = blockIdx.x * blockDim.x + threadIdx.x;
  if (t >= K * Ncols) return;
  int k = t / Ncols, c = t % Ncols;
  Wt[(size_t)c * K + k] = (_Float16)W[t];
}

// ---- WMMA GEMM: D[M,Ncols](f32) = A[M,K](f16 row-major) @ Bt[Ncols,K](f16) ----
// One wave32 per block computes a 16x64 strip (4 accumulators, A reused 4x).
// Requires M%16==0, K%32==0, Ncols==64 or Ncols%64==0.
__global__ void wmma_gemm4h(const _Float16* __restrict__ A,
                            const _Float16* __restrict__ Bt,
                            float* __restrict__ D, int K, int Ncols) {
  const int m0 = blockIdx.x * 16;
  const int n0 = blockIdx.y * 64;
  const int lane = threadIdx.x;          // 0..31
  const int r = lane & 15, g = lane >> 4;
  v8f cc[4] = {{}, {}, {}, {}};
  // A fragment: lane group g holds K-locals g*16 + j (contiguous)
  const _Float16* rowA = A + (size_t)(m0 + r) * K + g * 16;
  // B fragment: lane col = r(+tile), K-locals = g*16 + j (contiguous in Bt)
  const _Float16* colB[4];
#pragma unroll
  for (int nt = 0; nt < 4; ++nt)
    colB[nt] = Bt + (size_t)(n0 + nt * 16 + r) * K + g * 16;
  for (int kb = 0; kb < K; kb += 32) {
    if (kb + 32 < K) __builtin_prefetch(rowA + kb + 32, 0, 1);
    v8h alo = *(const v8h*)(rowA + kb);
    v8h ahi = *(const v8h*)(rowA + kb + 8);
    v16h a = __builtin_shufflevector(alo, ahi, 0, 1, 2, 3, 4, 5, 6, 7,
                                     8, 9, 10, 11, 12, 13, 14, 15);
#pragma unroll
    for (int nt = 0; nt < 4; ++nt) {
      v16h b = *(const v16h*)(colB[nt] + kb);
      cc[nt] = __builtin_amdgcn_wmma_f32_16x16x32_f16(false, a, false, b, (short)0,
                                                      cc[nt], false, false);
    }
  }
  // C/D: lane col = r, rows = g*8 + v
#pragma unroll
  for (int nt = 0; nt < 4; ++nt) {
    float* pd = D + (size_t)(m0 + g * 8) * Ncols + (n0 + nt * 16 + r);
#pragma unroll
    for (int v = 0; v < 8; ++v) pd[(size_t)v * Ncols] = cc[nt][v];
  }
}

// ---- per-node attention logits: al_src/al_dst [N,H] ----
__global__ void node_logits(const float* __restrict__ hfeat,
                            const float* __restrict__ att_s,
                            const float* __restrict__ att_d,
                            float* __restrict__ alsrc, float* __restrict__ aldst,
                            int N, int H, int Ch) {
  int t = blockIdx.x * blockDim.x + threadIdx.x;
  if (t >= N * H) return;
  int n = t / H, h = t % H;
  const float* p = hfeat + (size_t)n * H * Ch + (size_t)h * Ch;
  float s = 0.f, d = 0.f;
  for (int c = 0; c < Ch; ++c) {
    float v = p[c];
    s += v * att_s[h * Ch + c];
    d += v * att_d[h * Ch + c];
  }
  alsrc[t] = s;
  aldst[t] = d;
}

__device__ __forceinline__ void edge_sd(const int* ei, long long E, int e,
                                        int& s, int& d) {
  if (e < E) { s = ei[e]; d = ei[E + e]; }
  else       { s = d = e - (int)E; }
}

// ---- segment max over incoming edges (ordered-uint atomicMax) ----
__global__ void edge_max(const int* __restrict__ ei, long long E, int N, int H,
                         const float* __restrict__ alsrc,
                         const float* __restrict__ aldst,
                         unsigned* __restrict__ mmax) {
  long long t = (long long)blockIdx.x * blockDim.x + threadIdx.x;
  long long total = (E + N) * (long long)H;
  if (t >= total) return;
  int e = (int)(t / H), h = (int)(t % H);
  int s, d; edge_sd(ei, E, e, s, d);
  float lg = alsrc[s * H + h] + aldst[d * H + h];
  lg = lg > 0.f ? lg : NEG_SLOPE * lg;
  atomicMax(&mmax[d * H + h], f2ord(lg));
}

// ---- numerator: ealpha = exp(logit - max); denom[dst] += ealpha ----
__global__ void edge_expnum(const int* __restrict__ ei, long long E, int N, int H,
                            const float* __restrict__ alsrc,
                            const float* __restrict__ aldst,
                            const unsigned* __restrict__ mmax,
                            float* __restrict__ ealpha,
                            float* __restrict__ denom) {
  long long t = (long long)blockIdx.x * blockDim.x + threadIdx.x;
  long long total = (E + N) * (long long)H;
  if (t >= total) return;
  int e = (int)(t / H), h = (int)(t % H);
  int s, d; edge_sd(ei, E, e, s, d);
  float lg = alsrc[s * H + h] + aldst[d * H + h];
  lg = lg > 0.f ? lg : NEG_SLOPE * lg;
  float ex = expf(lg - ord2f(mmax[d * H + h]));
  ealpha[t] = ex;
  atomicAdd(&denom[d * H + h], ex);
}

// ---- normalize: ealpha /= denom[dst] ----
__global__ void edge_alpha(const int* __restrict__ ei, long long E, int N, int H,
                           const float* __restrict__ denom,
                           float* __restrict__ ealpha) {
  long long t = (long long)blockIdx.x * blockDim.x + threadIdx.x;
  long long total = (E + N) * (long long)H;
  if (t >= total) return;
  int e = (int)(t / H), h = (int)(t % H);
  int s, d; edge_sd(ei, E, e, s, d);
  ealpha[t] = ealpha[t] / denom[d * H + h];
}

// ---- scatter messages: agg[dst, :] += h[src, :] * alpha; 4 channels/thread ----
__global__ void edge_msg4(const int* __restrict__ ei, long long E, int N, int H,
                          int HC,
                          const float* __restrict__ ealpha,
                          const float* __restrict__ hfeat,
                          float* __restrict__ agg) {
  long long t = (long long)blockIdx.x * blockDim.x + threadIdx.x;
  const int qpe = HC >> 2;               // float4 quads per edge
  long long total = (E + N) * (long long)qpe;
  if (t >= total) return;
  int e = (int)(t / qpe);
  int idx = (int)(t % qpe) << 2;         // channel offset, multiple of 4
  int h = idx / (HC / H);
  int s, d; edge_sd(ei, E, e, s, d);
  float alpha = ealpha[(size_t)e * H + h];
  const float4 v = *(const float4*)(hfeat + (size_t)s * HC + idx);
  float* ag = agg + (size_t)d * HC + idx;
  atomicAdd(ag + 0, v.x * alpha);
  atomicAdd(ag + 1, v.y * alpha);
  atomicAdd(ag + 2, v.z * alpha);
  atomicAdd(ag + 3, v.w * alpha);
}

// ---- bias + LayerNorm + ReLU; one block (C threads) per node ----
// Optionally also writes an f16 copy (fragment-ready activation for next GEMM).
__global__ void ln_relu(const float* __restrict__ in, const float* __restrict__ bias,
                        const float* __restrict__ g, const float* __restrict__ b,
                        float* __restrict__ out, _Float16* __restrict__ outh, int C) {
  __shared__ float sred[256];
  int n = blockIdx.x, i = threadIdx.x;
  float v = in[(size_t)n * C + i] + bias[i];
  sred[i] = v;
  __syncthreads();
  for (int s = C >> 1; s > 0; s >>= 1) {
    if (i < s) sred[i] += sred[i + s];
    __syncthreads();
  }
  float mu = sred[0] / (float)C;
  __syncthreads();
  float dd = v - mu;
  sred[i] = dd * dd;
  __syncthreads();
  for (int s = C >> 1; s > 0; s >>= 1) {
    if (i < s) sred[i] += sred[i + s];
    __syncthreads();
  }
  float var = sred[0] / (float)C;
  float y = dd * rsqrtf(var + LN_EPS) * g[i] + b[i];
  float rel = fmaxf(y, 0.f);
  out[(size_t)n * C + i] = rel;
  if (outh) outh[(size_t)n * C + i] = (_Float16)rel;
}

// ---- MLP head: relu(h@fc1+b1)@fc2+b2, fc weights staged in LDS ----
__global__ void mlp_head(const float* __restrict__ hln2,
                         const float* __restrict__ fc1W, const float* __restrict__ fc1b,
                         const float* __restrict__ fc2W, const float* __restrict__ fc2b,
                         float* __restrict__ out, int N) {
  __shared__ float sW1[64 * 32];
  __shared__ float sW2[32];
  __shared__ float sb1[32];
  for (int i = threadIdx.x; i < 64 * 32; i += blockDim.x) sW1[i] = fc1W[i];
  if (threadIdx.x < 32) {
    sW2[threadIdx.x] = fc2W[threadIdx.x];
    sb1[threadIdx.x] = fc1b[threadIdx.x];
  }
  __syncthreads();
  int n = blockIdx.x * blockDim.x + threadIdx.x;
  if (n >= N) return;
  float xr[64];
#pragma unroll
  for (int k = 0; k < 64; ++k) xr[k] = hln2[(size_t)n * 64 + k];
  float acc = fc2b[0];
  for (int j = 0; j < 32; ++j) {
    float tv = sb1[j];
#pragma unroll
    for (int k = 0; k < 64; ++k) tv += xr[k] * sW1[k * 32 + j];
    acc += fmaxf(tv, 0.f) * sW2[j];
  }
  out[n] = acc;
}

extern "C" void kernel_launch(void* const* d_in, const int* in_sizes, int n_in,
                              void* d_out, int out_size, void* d_ws, size_t ws_size,
                              hipStream_t stream) {
  const float* x    = (const float*)d_in[0];
  const int*   ei   = (const int*)d_in[1];
  const float* W1   = (const float*)d_in[2];
  const float* as1  = (const float*)d_in[3];
  const float* ad1  = (const float*)d_in[4];
  const float* b1   = (const float*)d_in[5];
  const float* g1   = (const float*)d_in[6];
  const float* bb1  = (const float*)d_in[7];
  const float* W2   = (const float*)d_in[8];
  const float* as2  = (const float*)d_in[9];
  const float* ad2  = (const float*)d_in[10];
  const float* b2   = (const float*)d_in[11];
  const float* g2   = (const float*)d_in[12];
  const float* bb2  = (const float*)d_in[13];
  const float* fc1W = (const float*)d_in[14];
  const float* fc1b = (const float*)d_in[15];
  const float* fc2W = (const float*)d_in[16];
  const float* fc2b = (const float*)d_in[17];

  const int Fin = 128, H1 = 4, Ch = 64, HC1 = H1 * Ch; // 256
  const int N = in_sizes[0] / Fin;                     // 50000
  const long long E = in_sizes[1] / 2;                 // 800000
  const long long Etot = E + N;

  // workspace layout
  char* ws = (char*)d_ws;
  size_t off = 0;
  auto take = [&](size_t bytes) -> char* {
    char* p = ws + off;
    off += (bytes + 255) & ~(size_t)255;
    return p;
  };
  float* h1      = (float*)take((size_t)N * HC1 * 4); // h1, later hln1 (f32)
  float* big2    = (float*)take((size_t)N * HC1 * 4); // agg1, later h2/agg2/hln2
  float* alsrc   = (float*)take((size_t)N * H1 * 4);
  float* aldst   = (float*)take((size_t)N * H1 * 4);
  unsigned* mmax = (unsigned*)take((size_t)N * H1 * 4);
  float* denom   = (float*)take((size_t)N * H1 * 4);
  float* ealpha  = (float*)take((size_t)Etot * H1 * 4);
  _Float16* xh     = (_Float16*)take((size_t)N * Fin * 2);  // x in f16
  _Float16* hln1h  = (_Float16*)take((size_t)N * HC1 * 2);  // hln1 in f16
  _Float16* W1t    = (_Float16*)take((size_t)Fin * HC1 * 2); // W1^T f16 [HC1,Fin]
  _Float16* W2t    = (_Float16*)take((size_t)HC1 * 64 * 2);  // W2^T f16 [64,HC1]
  (void)ws_size; (void)n_in; (void)out_size;

  const unsigned NEG_INF_ORD = 0x007FFFFFu; // f2ord(-inf)
  auto blks = [](long long n, int bs) { return (unsigned)((n + bs - 1) / bs); };

  // ---- operand staging (f16 / transposed) ----
  cvt_f16<<<blks((long long)N * Fin / 4, 256), 256, 0, stream>>>(x, xh, (long long)N * Fin);
  cvt_transpose_f16<<<blks((long long)Fin * HC1, 256), 256, 0, stream>>>(W1, W1t, Fin, HC1);
  cvt_transpose_f16<<<blks((long long)HC1 * 64, 256), 256, 0, stream>>>(W2, W2t, HC1, 64);

  // ---- Layer 1 (H=4, C=64) ----
  wmma_gemm4h<<<dim3(N / 16, HC1 / 64), 32, 0, stream>>>(xh, W1t, h1, Fin, HC1);
  node_logits<<<blks((long long)N * H1, 256), 256, 0, stream>>>(
      h1, as1, ad1, alsrc, aldst, N, H1, Ch);
  fill_u32<<<blks((long long)N * H1, 256), 256, 0, stream>>>(mmax, (long long)N * H1, NEG_INF_ORD);
  fill_f32<<<blks((long long)N * H1, 256), 256, 0, stream>>>(denom, (long long)N * H1, 0.f);
  edge_max<<<blks(Etot * H1, 256), 256, 0, stream>>>(ei, E, N, H1, alsrc, aldst, mmax);
  edge_expnum<<<blks(Etot * H1, 256), 256, 0, stream>>>(ei, E, N, H1, alsrc, aldst,
                                                        mmax, ealpha, denom);
  edge_alpha<<<blks(Etot * H1, 256), 256, 0, stream>>>(ei, E, N, H1, denom, ealpha);
  fill_f32<<<blks((long long)N * HC1, 256), 256, 0, stream>>>(big2, (long long)N * HC1, 0.f);
  edge_msg4<<<blks(Etot * (HC1 / 4), 256), 256, 0, stream>>>(
      ei, E, N, H1, HC1, ealpha, h1, big2);
  ln_relu<<<N, HC1, 0, stream>>>(big2, b1, g1, bb1, h1, hln1h, HC1); // hln1 -> h1/hln1h

  // ---- Layer 2 (H=1, C=64) ----
  float* h2   = big2;                         // [N,64]
  float* agg2 = big2 + (size_t)N * 64;
  float* hln2 = big2 + (size_t)2 * N * 64;
  wmma_gemm4h<<<dim3(N / 16, 1), 32, 0, stream>>>(hln1h, W2t, h2, HC1, 64);
  node_logits<<<blks((long long)N, 256), 256, 0, stream>>>(
      h2, as2, ad2, alsrc, aldst, N, 1, Ch);
  fill_u32<<<blks((long long)N, 256), 256, 0, stream>>>(mmax, (long long)N, NEG_INF_ORD);
  fill_f32<<<blks((long long)N, 256), 256, 0, stream>>>(denom, (long long)N, 0.f);
  edge_max<<<blks(Etot, 256), 256, 0, stream>>>(ei, E, N, 1, alsrc, aldst, mmax);
  edge_expnum<<<blks(Etot, 256), 256, 0, stream>>>(ei, E, N, 1, alsrc, aldst,
                                                   mmax, ealpha, denom);
  edge_alpha<<<blks(Etot, 256), 256, 0, stream>>>(ei, E, N, 1, denom, ealpha);
  fill_f32<<<blks((long long)N * 64, 256), 256, 0, stream>>>(agg2, (long long)N * 64, 0.f);
  edge_msg4<<<blks(Etot * 16, 256), 256, 0, stream>>>(
      ei, E, N, 1, 64, ealpha, h2, agg2);
  ln_relu<<<N, 64, 0, stream>>>(agg2, b2, g2, bb2, hln2, (_Float16*)nullptr, 64);

  // ---- MLP head ----
  mlp_head<<<blks((long long)N, 128), 128, 0, stream>>>(
      hln2, fc1W, fc1b, fc2W, fc2b, (float*)d_out, N);
}
